// CircFormer_62457414419003
// MI455X (gfx1250) — matrix operand order repaired and checked by reading, not compiled, not run on hardware
//
#include <hip/hip_runtime.h>
#include <hip/hip_bf16.h>

typedef __attribute__((ext_vector_type(16))) _Float16 v16h;
typedef __attribute__((ext_vector_type(8)))  float    v8f;

#define NB   16
#define SL   5001
#define NC   128
#define NH   8
#define NDH  16
#define NM   44
#define NRES 8
#define NLAY 4
#define SC   (SL*NC)          // per-batch activation floats
#define RTOT (NB*SL)          // total rows
#define DN   0.5f             // 16^-0.25
#define RATIO 0.15075567228888181f  // 44^-0.5
#define NBPART 240

// ---------------------------------------------------------------------------
// A fragment (16x32 f16; this code's contiguous k order, same for B): lane holds row (lane&15);
// half = lane>>4 selects K range [16h, 16h+16).
// ---------------------------------------------------------------------------
__device__ __forceinline__ v16h load_a_frag(const float* ap, bool valid) {
  v16h a;
  if (valid) {
    float4 f0 = *(const float4*)(ap);
    float4 f1 = *(const float4*)(ap + 4);
    float4 f2 = *(const float4*)(ap + 8);
    float4 f3 = *(const float4*)(ap + 12);
    a[0]=(_Float16)f0.x; a[1]=(_Float16)f0.y; a[2]=(_Float16)f0.z; a[3]=(_Float16)f0.w;
    a[4]=(_Float16)f1.x; a[5]=(_Float16)f1.y; a[6]=(_Float16)f1.z; a[7]=(_Float16)f1.w;
    a[8]=(_Float16)f2.x; a[9]=(_Float16)f2.y; a[10]=(_Float16)f2.z; a[11]=(_Float16)f2.w;
    a[12]=(_Float16)f3.x; a[13]=(_Float16)f3.y; a[14]=(_Float16)f3.z; a[15]=(_Float16)f3.w;
  } else {
    for (int e = 0; e < 16; ++e) a[e] = (_Float16)0.f;
  }
  return a;
}

// ---------------------------------------------------------------------------
// WMMA GEMM:  Y[r,n] = sum_k A[r,k] * W[n,k]  (+bias[n]) (+=Y if ACC)
// Each wave computes 16 rows x 32 cols (two 16x16 WMMA tiles share the A
// fragment -> 1 A load/convert feeds 2 v_wmma).  4 waves/block = 128 cols.
// CONV mode: A is im2row of [S,128] activations, k = tap*128 + ci, K=384,
//            grid.z = batch; tap loop hoists the row-validity EXEC setup.
// W is f16 [N,K] row-major.
// ---------------------------------------------------------------------------
template<bool CONV, bool ACC>
__global__ __launch_bounds__(128) void gemm_wmma_k(
    const float* __restrict__ X, const _Float16* __restrict__ W,
    const float* __restrict__ bias, float* __restrict__ Y,
    int R, int K, int N)
{
  const int lane = threadIdx.x & 31;
  const int wave = threadIdx.x >> 5;
  const int nm   = lane & 15;       // A-row / B-col / C-col within tile
  const int half = lane >> 4;
  const int rowTile = blockIdx.x;
  const int colBase = (blockIdx.y * 4 + wave) * 32;

  const float* Xb = CONV ? (X + (size_t)blockIdx.z * R * NC) : X;
  float*       Yb = CONV ? (Y + (size_t)blockIdx.z * R * N)  : Y;
  const int KK = CONV ? 3 * NC : K;

  const _Float16* w0 = W + (size_t)(colBase + nm) * KK + half * 16;
  const _Float16* w1 = W + (size_t)(colBase + 16 + nm) * KK + half * 16;

  v8f acc0 = {0.f,0.f,0.f,0.f,0.f,0.f,0.f,0.f};
  v8f acc1 = {0.f,0.f,0.f,0.f,0.f,0.f,0.f,0.f};

  if (CONV) {
    for (int tap = 0; tap < 3; ++tap) {
      int srow = rowTile * 16 + nm + tap - 1;
      bool avalid = (srow >= 0) && (srow < R);
      const float* arow = Xb + (size_t)(avalid ? srow : 0) * NC + half * 16;
      // prefetch the A row one row-tile ahead (streaming hint)
      __builtin_prefetch((const void*)(arow + 16 * NC), 0, 1);
      for (int ci0 = 0; ci0 < NC; ci0 += 32) {
        v16h a = load_a_frag(arow + ci0, avalid);
        int kc = tap * NC + ci0;
        v16h b0 = *(const v16h*)(w0 + kc);
        v16h b1 = *(const v16h*)(w1 + kc);
        acc0 = __builtin_amdgcn_wmma_f32_16x16x32_f16(false, a, false, b0,
                                                      (short)0, acc0, false, false);
        acc1 = __builtin_amdgcn_wmma_f32_16x16x32_f16(false, a, false, b1,
                                                      (short)0, acc1, false, false);
      }
    }
  } else {
    int row = rowTile * 16 + nm;
    bool avalid = row < R;
    const float* arow = Xb + (size_t)(avalid ? row : 0) * K + half * 16;
    __builtin_prefetch((const void*)(arow + (size_t)16 * K), 0, 1);
    for (int kc = 0; kc < KK; kc += 32) {
      v16h a = load_a_frag(arow + kc, avalid);
      v16h b0 = *(const v16h*)(w0 + kc);
      v16h b1 = *(const v16h*)(w1 + kc);
      acc0 = __builtin_amdgcn_wmma_f32_16x16x32_f16(false, a, false, b0,
                                                    (short)0, acc0, false, false);
      acc1 = __builtin_amdgcn_wmma_f32_16x16x32_f16(false, a, false, b1,
                                                    (short)0, acc1, false, false);
    }
  }

  // ---- epilogue: VGPR r -> M = 8*half + r, cols = colBase+nm, +16
  int col0 = colBase + nm;
  int col1 = col0 + 16;
  float bv0 = bias ? bias[col0] : 0.f;
  float bv1 = bias ? bias[col1] : 0.f;
  for (int r = 0; r < 8; ++r) {
    int row = rowTile * 16 + half * 8 + r;
    if (row < R) {
      size_t idx0 = (size_t)row * N + col0;
      size_t idx1 = (size_t)row * N + col1;
      float v0 = acc0[r] + bv0;
      float v1 = acc1[r] + bv1;
      if (ACC) { v0 += Yb[idx0]; v1 += Yb[idx1]; }
      Yb[idx0] = v0;
      Yb[idx1] = v1;
    }
  }
}

// ---------------------------------------------------------------------------
// Weight packing
// ---------------------------------------------------------------------------
__global__ void cvt_k(const float* __restrict__ s, _Float16* __restrict__ d, size_t n) {
  size_t i = (size_t)blockIdx.x * 256 + threadIdx.x;
  if (i < n) d[i] = (_Float16)s[i];
}
// res weights [8][co][ci][tap] -> [8][co][tap*128+ci] f16
__global__ void pack_res_k(const float* __restrict__ s, _Float16* __restrict__ d) {
  size_t i = (size_t)blockIdx.x * 256 + threadIdx.x;
  if (i >= (size_t)NRES * NC * 3 * NC) return;
  int k = (int)(i % (3 * NC));
  size_t rem = i / (3 * NC);
  int co = (int)(rem % NC);
  int li = (int)(rem / NC);
  int tap = k >> 7, ci = k & 127;
  d[i] = (_Float16)s[(((size_t)li * NC + co) * NC + ci) * 3 + tap];
}

// ---------------------------------------------------------------------------
// Front end
// ---------------------------------------------------------------------------
__global__ void conv1_k(const float* __restrict__ x, const float* __restrict__ w,
                        const float* __restrict__ bias, float* __restrict__ Y) {
  size_t gid = (size_t)blockIdx.x * 256 + threadIdx.x;
  if (gid >= (size_t)RTOT * NC) return;
  int co = gid & 127;
  size_t rs = gid >> 7;
  int s = (int)(rs % SL);
  size_t b = rs / SL;
  float acc = bias[co];
  for (int t = 0; t < 3; ++t) {
    int sp = s + t - 1;
    if (sp >= 0 && sp < SL) {
      const float* xr = x + ((size_t)b * SL + sp) * 4;
      for (int ci = 0; ci < 4; ++ci) acc += xr[ci] * w[(co * 4 + ci) * 3 + t];
    }
  }
  Y[gid] = acc;
}

__global__ __launch_bounds__(128) void bn_part_k(const float* __restrict__ Y,
                                                 float* __restrict__ part, int rows) {
  int c = threadIdx.x;
  float s = 0.f, sq = 0.f;
  for (int r = blockIdx.x; r < rows; r += gridDim.x) {
    float v = Y[(size_t)r * NC + c];
    s += v; sq += v * v;
  }
  part[(size_t)blockIdx.x * 256 + c] = s;
  part[(size_t)blockIdx.x * 256 + 128 + c] = sq;
}
__global__ __launch_bounds__(128) void bn_fin_k(const float* __restrict__ part,
                                                const float* __restrict__ g,
                                                const float* __restrict__ be,
                                                float* __restrict__ ss, int rows, int nb) {
  int c = threadIdx.x;
  float s = 0.f, sq = 0.f;
  for (int i = 0; i < nb; ++i) { s += part[(size_t)i * 256 + c]; sq += part[(size_t)i * 256 + 128 + c]; }
  float inv = 1.f / (float)rows;
  float mean = s * inv;
  float var = sq * inv - mean * mean;
  float scale = g[c] * rsqrtf(var + 1e-5f);
  ss[c] = scale;
  ss[128 + c] = be[c] - mean * scale;
}
__global__ void bn_relu_k(const float* __restrict__ Y, const float* __restrict__ ss,
                          float* __restrict__ H, size_t n) {
  size_t i = (size_t)blockIdx.x * 256 + threadIdx.x;
  if (i >= n) return;
  int c = i & 127;
  H[i] = fmaxf(Y[i] * ss[c] + ss[128 + c], 0.f);
}
__global__ void bn_add_relu_k(const float* __restrict__ Y, const float* __restrict__ ss,
                              float* __restrict__ H, size_t n) {
  size_t i = (size_t)blockIdx.x * 256 + threadIdx.x;
  if (i >= n) return;
  int c = i & 127;
  H[i] = fmaxf(Y[i] * ss[c] + ss[128 + c] + H[i], 0.f);
}

__global__ void pe_k(float* __restrict__ H) {
  size_t i = (size_t)blockIdx.x * 256 + threadIdx.x;
  if (i >= (size_t)RTOT * NC) return;
  int c = i & 127;
  int s = (int)((i >> 7) % SL);
  float div = expf((float)(c & ~1) * (-9.2103403719761836f / 128.f)); // -ln(1e4)/C
  float ang = (float)s * div;
  H[i] += (c & 1) ? cosf(ang) : sinf(ang);
}

// ---------------------------------------------------------------------------
// LayerNorm / GELU / head: wave-per-row reductions (wave32)
// ---------------------------------------------------------------------------
__global__ __launch_bounds__(256) void ln_k(const float* __restrict__ X,
                                            const float* __restrict__ g,
                                            const float* __restrict__ b,
                                            float* __restrict__ Y, int rows) {
  int gid = blockIdx.x * 256 + threadIdx.x;
  int row = gid >> 5, lane = threadIdx.x & 31;
  if (row >= rows) return;
  const float* x = X + (size_t)row * NC;
  float v[4]; float s = 0.f, sq = 0.f;
  for (int j = 0; j < 4; ++j) { v[j] = x[lane + 32 * j]; s += v[j]; sq += v[j] * v[j]; }
  for (int o = 16; o > 0; o >>= 1) { s += __shfl_xor(s, o, 32); sq += __shfl_xor(sq, o, 32); }
  float mean = s * (1.f / 128.f);
  float var = sq * (1.f / 128.f) - mean * mean;
  float rstd = rsqrtf(var + 1e-5f);
  float* y = Y + (size_t)row * NC;
  for (int j = 0; j < 4; ++j) { int c = lane + 32 * j; y[c] = (v[j] - mean) * rstd * g[c] + b[c]; }
}

__global__ void gelu_k(float* __restrict__ Z, size_t n) {
  size_t i = (size_t)blockIdx.x * 256 + threadIdx.x;
  if (i >= n) return;
  float x = Z[i];
  Z[i] = 0.5f * x * (1.f + erff(x * 0.70710678118654752f));
}

__global__ __launch_bounds__(256) void head_k(const float* __restrict__ X,
                                              const float* __restrict__ hw,
                                              const float* __restrict__ hb,
                                              const int* __restrict__ sid,
                                              float* __restrict__ out) {
  int gid = blockIdx.x * 256 + threadIdx.x;
  int row = gid >> 5, lane = threadIdx.x & 31;
  if (row >= RTOT) return;
  int id = sid[row / SL];
  const float* x = X + (size_t)row * NC;
  const float* w = hw + (size_t)id * NC;
  float s = 0.f;
  for (int j = 0; j < 4; ++j) { int c = lane + 32 * j; s += x[c] * w[c]; }
  for (int o = 16; o > 0; o >>= 1) s += __shfl_xor(s, o, 32);
  if (lane == 0) out[row] = s + hb[id];
}

// ---------------------------------------------------------------------------
// FAVOR+ attention (no qp/kp materialization)
// ---------------------------------------------------------------------------
__global__ __launch_bounds__(256) void kmax_part_k(const float* __restrict__ Kx,
                                                   const float* __restrict__ proj,
                                                   float* __restrict__ part) {
  __shared__ float sProj[NM][16];
  __shared__ float red[256];
  int t = threadIdx.x;
  for (int i = t; i < NM * 16; i += 256) sProj[i >> 4][i & 15] = proj[i];
  __syncthreads();
  const size_t ROWS = (size_t)RTOT * NH;
  float mx = -3.4e38f;
  for (size_t r = (size_t)blockIdx.x * 256 + t; r < ROWS; r += (size_t)gridDim.x * 256) {
    int hh = (int)(r & 7);
    size_t bs = r >> 3;
    const float* kp = Kx + bs * NC + hh * NDH;
    float k[16];
    for (int e = 0; e < 16; ++e) k[e] = kp[e];
    for (int m = 0; m < NM; ++m) {
      float s = 0.f;
      for (int e = 0; e < 16; ++e) s += k[e] * sProj[m][e];
      mx = fmaxf(mx, DN * s);
    }
  }
  red[t] = mx; __syncthreads();
  for (int o = 128; o > 0; o >>= 1) { if (t < o) red[t] = fmaxf(red[t], red[t + o]); __syncthreads(); }
  if (t == 0) part[blockIdx.x] = red[0];
}
__global__ void kmax_fin_k(const float* __restrict__ part, float* __restrict__ kmax, int n) {
  __shared__ float red[256];
  int t = threadIdx.x;
  float mx = -3.4e38f;
  for (int i = t; i < n; i += 256) mx = fmaxf(mx, part[i]);
  red[t] = mx; __syncthreads();
  for (int o = 128; o > 0; o >>= 1) { if (t < o) red[t] = fmaxf(red[t], red[t + o]); __syncthreads(); }
  if (t == 0) *kmax = red[0];
}

// block per (b,h): ctx[m,d] = sum_s kp[s,m] v[s,d];  ksum[m] = sum_s kp[s,m]
__global__ __launch_bounds__(256) void favor_ctx_k(const float* __restrict__ Kx,
                                                   const float* __restrict__ Vx,
                                                   const float* __restrict__ proj,
                                                   const float* __restrict__ kmax,
                                                   float* __restrict__ ctx,
                                                   float* __restrict__ ksum) {
  __shared__ float sProj[NM][16];
  __shared__ float sK[16][16];
  __shared__ float sV[16][16];
  __shared__ float sKP[16][46];
  const int t = threadIdx.x;
  const int bh = blockIdx.x;
  const int b = bh >> 3, hh = bh & 7;
  for (int i = t; i < NM * 16; i += 256) sProj[i >> 4][i & 15] = proj[i];
  const float KM = *kmax;
  const int d = t & 15;
  const int mb = (t >> 4) * 3;
  float accC[3] = {0.f, 0.f, 0.f};
  float accS[3] = {0.f, 0.f, 0.f};
  for (int s0 = 0; s0 < SL; s0 += 16) {
    __syncthreads();
    {
      int sl = t >> 4, c = t & 15;
      int s = s0 + sl;
      float kv = 0.f, vv = 0.f;
      if (s < SL) {
        size_t base = ((size_t)b * SL + s) * NC + hh * NDH + c;
        kv = Kx[base]; vv = Vx[base];
      }
      sK[sl][c] = kv; sV[sl][c] = vv;
    }
    __syncthreads();
    for (int idx = t; idx < 16 * NM; idx += 256) {
      int sl = idx / NM, m = idx % NM;
      float kp = 0.f;
      if (s0 + sl < SL) {
        float dd = 0.f, sq = 0.f;
        for (int e = 0; e < 16; ++e) {
          float kv = sK[sl][e];
          dd += kv * sProj[m][e];
          sq += kv * kv;
        }
        kp = RATIO * (expf(DN * dd - 0.125f * sq - KM) + 1e-4f);
      }
      sKP[sl][m] = kp;
    }
    __syncthreads();
    for (int j = 0; j < 3; ++j) {
      int m = mb + j;
      if (m < NM) {
        float ca = 0.f, sa = 0.f;
        for (int sl = 0; sl < 16; ++sl) {
          float kp = sKP[sl][m];
          ca += kp * sV[sl][d];
          sa += kp;
        }
        accC[j] += ca;
        accS[j] += sa;
      }
    }
  }
  for (int j = 0; j < 3; ++j) {
    int m = mb + j;
    if (m < NM) {
      ctx[((size_t)bh * NM + m) * 16 + d] = accC[j];
      if (d == 0) ksum[(size_t)bh * NM + m] = accS[j];
    }
  }
}

// thread per (b,s,h): att = (qp . ctx) / (qp . ksum)
__global__ __launch_bounds__(256) void favor_out_k(const float* __restrict__ Qx,
                                                   const float* __restrict__ proj,
                                                   const float* __restrict__ ctx,
                                                   const float* __restrict__ ksum,
                                                   float* __restrict__ att) {
  __shared__ float sProj[NM][16];
  int t = threadIdx.x;
  for (int i = t; i < NM * 16; i += 256) sProj[i >> 4][i & 15] = proj[i];
  __syncthreads();
  size_t r = (size_t)blockIdx.x * 256 + t;
  if (r >= (size_t)RTOT * NH) return;
  int hh = (int)(r & 7);
  size_t bs = r >> 3;
  const float* qp = Qx + bs * NC + hh * NDH;
  float q[16]; float sq = 0.f;
  for (int e = 0; e < 16; ++e) { q[e] = qp[e]; sq += q[e] * q[e]; }
  float dd[NM]; float mx = -3.4e38f;
  for (int m = 0; m < NM; ++m) {
    float s = 0.f;
    for (int e = 0; e < 16; ++e) s += q[e] * sProj[m][e];
    s *= DN;
    dd[m] = s;
    mx = fmaxf(mx, s);
  }
  float diag = 0.125f * sq;
  size_t bh = (bs / SL) * NH + hh;
  const float* cx = ctx + bh * NM * 16;
  const float* ks = ksum + bh * NM;
  float acc[16];
  for (int e = 0; e < 16; ++e) acc[e] = 0.f;
  float den = 0.f;
  for (int m = 0; m < NM; ++m) {
    float qv = RATIO * (expf(dd[m] - diag - mx) + 1e-4f);
    den += qv * ks[m];
    const float* cr = cx + m * 16;
    for (int e = 0; e < 16; ++e) acc[e] += qv * cr[e];
  }
  float dinv = 1.f / den;
  float* op = att + bs * NC + hh * NDH;
  for (int e = 0; e < 16; ++e) op[e] = acc[e] * dinv;
}

// ---------------------------------------------------------------------------
extern "C" void kernel_launch(void* const* d_in, const int* in_sizes, int n_in,
                              void* d_out, int out_size, void* d_ws, size_t ws_size,
                              hipStream_t stream) {
  const float* x        = (const float*)d_in[0];
  const float* conv1_w  = (const float*)d_in[1];
  const float* conv1_b  = (const float*)d_in[2];
  const float* bn0_g    = (const float*)d_in[3];
  const float* bn0_b    = (const float*)d_in[4];
  const float* res_w1   = (const float*)d_in[5];
  const float* res_b1   = (const float*)d_in[6];
  const float* res_g1   = (const float*)d_in[7];
  const float* res_be1  = (const float*)d_in[8];
  const float* res_w2   = (const float*)d_in[9];
  const float* res_b2   = (const float*)d_in[10];
  const float* res_g2   = (const float*)d_in[11];
  const float* res_be2  = (const float*)d_in[12];
  const float* ln1_g    = (const float*)d_in[13];
  const float* ln1_b    = (const float*)d_in[14];
  const float* wq       = (const float*)d_in[15];
  const float* wk       = (const float*)d_in[16];
  const float* wv       = (const float*)d_in[17];
  const float* wo       = (const float*)d_in[18];
  const float* bq       = (const float*)d_in[19];
  const float* bk       = (const float*)d_in[20];
  const float* bv       = (const float*)d_in[21];
  const float* bo       = (const float*)d_in[22];
  const float* projA    = (const float*)d_in[23];
  const float* ln2_g    = (const float*)d_in[24];
  const float* ln2_b    = (const float*)d_in[25];
  const float* ff_w1    = (const float*)d_in[26];
  const float* ff_b1    = (const float*)d_in[27];
  const float* ff_w2    = (const float*)d_in[28];
  const float* ff_b2    = (const float*)d_in[29];
  const float* final_w  = (const float*)d_in[30];
  const float* final_b  = (const float*)d_in[31];
  const float* head_w   = (const float*)d_in[32];
  const float* head_b   = (const float*)d_in[33];
  const int*   sid      = (const int*)d_in[34];
  float* out = (float*)d_out;

  // ---- workspace carve (256B aligned) --------------------------------------
  char* p = (char*)d_ws;
  auto carve = [&](size_t bytes) -> void* {
    void* r = (void*)p;
    p += (bytes + 255) & ~(size_t)255;
    return r;
  };
  const size_t RC = (size_t)RTOT * NC;
  float* H   = (float*)carve(RC * 4);
  float* T1  = (float*)carve(RC * 4);
  float* T2  = (float*)carve(RC * 4);
  float* T3  = (float*)carve(RC * 4);
  float* T4  = (float*)carve(RC * 4);
  float* Z   = (float*)carve((size_t)SL * 512 * 4);
  float* CTX = (float*)carve((size_t)NB * NH * NM * 16 * 4);
  float* KSM = (float*)carve((size_t)NB * NH * NM * 4);
  float* SS  = (float*)carve(256 * 4);
  float* PART  = (float*)carve((size_t)NBPART * 256 * 4);
  float* KPART = (float*)carve(256 * 4);
  float* KMAX  = (float*)carve(4);
  _Float16* WR1  = (_Float16*)carve((size_t)NRES * NC * 384 * 2);
  _Float16* WR2  = (_Float16*)carve((size_t)NRES * NC * 384 * 2);
  _Float16* WQ   = (_Float16*)carve((size_t)NLAY * NC * NC * 2);
  _Float16* WK   = (_Float16*)carve((size_t)NLAY * NC * NC * 2);
  _Float16* WV   = (_Float16*)carve((size_t)NLAY * NC * NC * 2);
  _Float16* WO   = (_Float16*)carve((size_t)NLAY * NC * NC * 2);
  _Float16* WF1  = (_Float16*)carve((size_t)NLAY * 512 * NC * 2);
  _Float16* WF2  = (_Float16*)carve((size_t)NLAY * NC * 512 * 2);
  _Float16* WFIN = (_Float16*)carve((size_t)NC * NC * 2);

  auto blocks = [](size_t n) { return (unsigned)((n + 255) / 256); };

  // ---- pack weights to f16 -------------------------------------------------
  {
    size_t n = (size_t)NRES * NC * 384;
    pack_res_k<<<blocks(n), 256, 0, stream>>>(res_w1, WR1);
    pack_res_k<<<blocks(n), 256, 0, stream>>>(res_w2, WR2);
    size_t nq = (size_t)NLAY * NC * NC;
    cvt_k<<<blocks(nq), 256, 0, stream>>>(wq, WQ, nq);
    cvt_k<<<blocks(nq), 256, 0, stream>>>(wk, WK, nq);
    cvt_k<<<blocks(nq), 256, 0, stream>>>(wv, WV, nq);
    cvt_k<<<blocks(nq), 256, 0, stream>>>(wo, WO, nq);
    size_t nf = (size_t)NLAY * 512 * NC;
    cvt_k<<<blocks(nf), 256, 0, stream>>>(ff_w1, WF1, nf);
    cvt_k<<<blocks(nf), 256, 0, stream>>>(ff_w2, WF2, nf);
    cvt_k<<<blocks((size_t)NC * NC), 256, 0, stream>>>(final_w, WFIN, (size_t)NC * NC);
  }

  const int STILES = (SL + 15) / 16;  // 313

  // ---- stem: conv1 + bn + relu --------------------------------------------
  conv1_k<<<blocks(RC), 256, 0, stream>>>(x, conv1_w, conv1_b, T1);
  bn_part_k<<<NBPART, 128, 0, stream>>>(T1, PART, RTOT);
  bn_fin_k<<<1, 128, 0, stream>>>(PART, bn0_g, bn0_b, SS, RTOT, NBPART);
  bn_relu_k<<<blocks(RC), 256, 0, stream>>>(T1, SS, H, RC);

  // ---- residual blocks -----------------------------------------------------
  for (int i = 0; i < NRES; ++i) {
    gemm_wmma_k<true, false><<<dim3(STILES, 1, NB), 128, 0, stream>>>(
        H, WR1 + (size_t)i * NC * 384, res_b1 + i * NC, T1, SL, 384, NC);
    bn_part_k<<<NBPART, 128, 0, stream>>>(T1, PART, RTOT);
    bn_fin_k<<<1, 128, 0, stream>>>(PART, res_g1 + i * NC, res_be1 + i * NC, SS, RTOT, NBPART);
    bn_relu_k<<<blocks(RC), 256, 0, stream>>>(T1, SS, T1, RC);

    gemm_wmma_k<true, false><<<dim3(STILES, 1, NB), 128, 0, stream>>>(
        T1, WR2 + (size_t)i * NC * 384, res_b2 + i * NC, T2, SL, 384, NC);
    bn_part_k<<<NBPART, 128, 0, stream>>>(T2, PART, RTOT);
    bn_fin_k<<<1, 128, 0, stream>>>(PART, res_g2 + i * NC, res_be2 + i * NC, SS, RTOT, NBPART);
    bn_add_relu_k<<<blocks(RC), 256, 0, stream>>>(T2, SS, H, RC);
  }

  // ---- positional encoding -------------------------------------------------
  pe_k<<<blocks(RC), 256, 0, stream>>>(H);

  const unsigned LNG = blocks((size_t)RTOT * 32);
  const int RT_FULL = RTOT / 16;  // 5001, exact

  // ---- transformer layers --------------------------------------------------
  for (int l = 0; l < NLAY; ++l) {
    const float* proj = projA + (size_t)l * NM * NDH;
    // LN1 -> T1
    ln_k<<<LNG, 256, 0, stream>>>(H, ln1_g + l * NC, ln1_b + l * NC, T1, RTOT);
    // Q,K,V
    gemm_wmma_k<false, false><<<dim3(RT_FULL, 1), 128, 0, stream>>>(
        T1, WQ + (size_t)l * NC * NC, bq + l * NC, T2, RTOT, NC, NC);
    gemm_wmma_k<false, false><<<dim3(RT_FULL, 1), 128, 0, stream>>>(
        T1, WK + (size_t)l * NC * NC, bk + l * NC, T3, RTOT, NC, NC);
    gemm_wmma_k<false, false><<<dim3(RT_FULL, 1), 128, 0, stream>>>(
        T1, WV + (size_t)l * NC * NC, bv + l * NC, T4, RTOT, NC, NC);
    // global k feature max
    kmax_part_k<<<256, 256, 0, stream>>>(T3, proj, KPART);
    kmax_fin_k<<<1, 256, 0, stream>>>(KPART, KMAX, 256);
    // ctx / ksum, then attention output -> T1
    favor_ctx_k<<<NB * NH, 256, 0, stream>>>(T3, T4, proj, KMAX, CTX, KSM);
    favor_out_k<<<blocks((size_t)RTOT * NH), 256, 0, stream>>>(T2, proj, CTX, KSM, T1);
    // H += att @ Wo^T + bo
    gemm_wmma_k<false, true><<<dim3(RT_FULL, 1), 128, 0, stream>>>(
        T1, WO + (size_t)l * NC * NC, bo + l * NC, H, RTOT, NC, NC);
    // LN2 -> T1
    ln_k<<<LNG, 256, 0, stream>>>(H, ln2_g + l * NC, ln2_b + l * NC, T1, RTOT);
    // FF, chunked per batch element (Z scratch = [S,512])
    for (int b = 0; b < NB; ++b) {
      gemm_wmma_k<false, false><<<dim3(STILES, 4), 128, 0, stream>>>(
          T1 + (size_t)b * SC, WF1 + (size_t)l * 512 * NC, ff_b1 + l * 512, Z, SL, NC, 512);
      gelu_k<<<blocks((size_t)SL * 512), 256, 0, stream>>>(Z, (size_t)SL * 512);
      gemm_wmma_k<false, true><<<dim3(STILES, 1), 128, 0, stream>>>(
          Z, WF2 + (size_t)l * NC * 512, ff_b2 + l * NC, H + (size_t)b * SC, SL, 512, NC);
    }
  }

  // ---- final 1x1 conv + species head --------------------------------------
  gemm_wmma_k<false, false><<<dim3(RT_FULL, 1), 128, 0, stream>>>(
      H, WFIN, final_b, T1, RTOT, NC, NC);
  head_k<<<LNG, 256, 0, stream>>>(T1, head_w, head_b, sid, out);
}
